// SimpleGAT_26027501814347
// MI455X (gfx1250) — matrix operand order repaired and checked by reading, not compiled, not run on hardware
//
#include <hip/hip_runtime.h>
#include <hip/hip_bf16.h>
#include <stdint.h>

// ---------------- problem constants (from reference) ----------------
#define NN 50000
#define EE 800000
#define HH 4
#define GG 32
#define CC 4
#define ETOT (EE + NN)          // edges + self loops
#define NEG_SLOPE 0.2f
#define EPS_GN 1e-5f

typedef __attribute__((ext_vector_type(16))) __bf16 v16bf;
typedef __attribute__((ext_vector_type(8)))  float  v8f;

// ---------------- small device helpers ----------------
__device__ __forceinline__ unsigned short f2bf(float f) {
  unsigned u = __float_as_uint(f);
  u += 0x7fffu + ((u >> 16) & 1u);      // round-to-nearest-even
  return (unsigned short)(u >> 16);
}
// order-preserving float<->int mapping for atomicMax on signed ints
__device__ __forceinline__ int f2ord(float f) {
  int i = __float_as_int(f);
  return (i >= 0) ? i : (i ^ 0x7fffffff);
}
__device__ __forceinline__ float ord2f(int i) {
  return __int_as_float((i >= 0) ? i : (i ^ 0x7fffffff));
}

// ---------------- utility kernels ----------------
__global__ void fill_f32_kernel(float* p, float v, int n) {
  int t = blockIdx.x * blockDim.x + threadIdx.x;
  if (t < n) p[t] = v;
}
__global__ void fill_i32_kernel(int* p, int v, int n) {
  int t = blockIdx.x * blockDim.x + threadIdx.x;
  if (t < n) p[t] = v;
}
__global__ void bias_bcast_kernel(float* O, const float* __restrict__ bias, int mask, int n) {
  int t = blockIdx.x * blockDim.x + threadIdx.x;
  if (t < n) O[t] = bias[t & mask];
}
__global__ void count_nodes_kernel(const int* __restrict__ batch, float* cnt, int n) {
  int t = blockIdx.x * blockDim.x + threadIdx.x;
  if (t < n) atomicAdd(&cnt[batch[t]], 1.0f);
}
__global__ void clamp1_kernel(float* p, int n) {
  int t = blockIdx.x * blockDim.x + threadIdx.x;
  if (t < n) p[t] = fmaxf(p[t], 1.0f);
}

// ---------------- fp32 -> bf16 activation conversion (8 elems/thread) ----------------
__global__ void cvt_bf16_kernel(const float* __restrict__ X, unsigned short* __restrict__ Xb,
                                int n8) {
  int t = blockIdx.x * blockDim.x + threadIdx.x;
  if (t >= n8) return;
  const float4* p = (const float4*)(X + (size_t)t * 8);
  float4 a = p[0], b = p[1];
  union { unsigned short u[8]; uint4 q; } r;
  r.u[0] = f2bf(a.x); r.u[1] = f2bf(a.y); r.u[2] = f2bf(a.z); r.u[3] = f2bf(a.w);
  r.u[4] = f2bf(b.x); r.u[5] = f2bf(b.y); r.u[6] = f2bf(b.z); r.u[7] = f2bf(b.w);
  ((uint4*)Xb)[t] = r.q;
}

// ---------------- weight packing: fp32 [K,Nout] -> bf16 B-fragments ----------------
// Fragment for (ntile, kc, lane): 16 bf16 contiguous; element j holds
// W[kc*32 + (lane>>4)*16 + j][ntile*16 + (lane&15)]   (contiguous k order, same as the A loads)
__global__ void pack_w_kernel(const float* __restrict__ W, unsigned short* __restrict__ Wp,
                              int K, int Nout) {
  int tid = blockIdx.x * blockDim.x + threadIdx.x;
  int nK = K >> 5;
  int total = (Nout >> 4) * nK * 32;
  if (tid >= total) return;
  int lane  = tid & 31;
  int t     = tid >> 5;
  int kc    = t % nK;
  int ntile = t / nK;
  int col   = (ntile << 4) + (lane & 15);
  int kbase = (kc << 5) + ((lane >> 4) << 4);
  unsigned short* dst = Wp + (size_t)tid * 16;
#pragma unroll
  for (int j = 0; j < 16; ++j)
    dst[j] = f2bf(W[(size_t)(kbase + j) * Nout + col]);
}

// ---------------- bf16 WMMA GEMM:  OUT[NN,Nout] = Xb[NN,K] @ W + bias ----------------
// 4 waves/block, each wave owns one 16x16 tile (same M-tile, 4 N-tiles).
// K templated so the k-loop fully unrolls: loads clause up front, WMMAs back-to-back.
#define WPB 4

template <int K>
__global__ __launch_bounds__(128) void gat_gemm_bf16_wmma(
    const unsigned short* __restrict__ Xb, const unsigned short* __restrict__ Wp,
    const float* __restrict__ bias, float* __restrict__ OUT, int Nout) {
  constexpr int nK = K >> 5;
  const int lane  = threadIdx.x & 31;
  const int wave  = threadIdx.x >> 5;
  const int mtile = blockIdx.x;
  const int ntile = blockIdx.y * WPB + wave;

  const int m    = (mtile << 4) + (lane & 15);
  const int kblk = (lane >> 4) << 4;           // A K-block base: 0 or 16
  const unsigned short* xrow  = Xb + (size_t)m * K;
  const unsigned short* wbase = Wp + ((size_t)ntile * nK * 32 + lane) * 16;

  v8f acc = {0.f, 0.f, 0.f, 0.f, 0.f, 0.f, 0.f, 0.f};

#pragma unroll
  for (int kc = 0; kc < nK; ++kc) {
    const int kb = kc << 5;
    // A fragment: element j holds K = kblk + j -> two 16B loads
    union { v16bf v; uint4 q[2]; } A;
    A.q[0] = *(const uint4*)(xrow + kb + kblk);
    A.q[1] = *(const uint4*)(xrow + kb + kblk + 8);
    // B fragment: pre-packed, 32 contiguous bytes per lane
    union { v16bf v; uint4 q[2]; } B;
    const uint4* wq = (const uint4*)(wbase + (size_t)kc * 512);
    B.q[0] = wq[0]; B.q[1] = wq[1];

    acc = __builtin_amdgcn_wmma_f32_16x16x32_bf16(
        /*neg_a=*/false, A.v, /*neg_b=*/false, B.v,
        /*c_mod=*/(short)0, acc, /*reuse_a=*/false, /*reuse_b=*/false);
  }

  const int col   = (ntile << 4) + (lane & 15);
  const float bv  = bias[col];
  const int mbase = (mtile << 4) + ((lane >> 4) << 3);  // D row = r + (lane>>4)*8
#pragma unroll
  for (int r = 0; r < 8; ++r)
    OUT[(size_t)(mbase + r) * Nout + col] = acc[r] + bv;
}

// ---------------- edge phase (self-loops are edges E..E+N-1) ----------------
__global__ void edge_score_kernel(const float* __restrict__ xl, const float* __restrict__ xr,
                                  const float* __restrict__ att, const int* __restrict__ ei,
                                  float* __restrict__ s, int* __restrict__ smaxI,
                                  int D, int HD) {
  int tid = blockIdx.x * blockDim.x + threadIdx.x;
  if (tid >= ETOT * HH) return;
  int e = tid >> 2, h = tid & 3;
  int src, dst;
  if (e < EE) { src = ei[e]; dst = ei[EE + e]; } else { src = dst = e - EE; }
  const float* pl = xl + (size_t)src * HD + h * D;
  const float* pr = xr + (size_t)dst * HD + h * D;
  const float* pa = att + h * D;
  float acc = 0.f;
  for (int d = 0; d < D; ++d) {
    float m = pl[d] + pr[d];
    m = (m > 0.f) ? m : NEG_SLOPE * m;
    acc += m * pa[d];
  }
  s[tid] = acc;
  atomicMax(smaxI + dst * HH + h, f2ord(acc));
}

__global__ void edge_softmax_kernel(const int* __restrict__ ei, const int* __restrict__ smaxI,
                                    float* __restrict__ s, float* __restrict__ denom) {
  int tid = blockIdx.x * blockDim.x + threadIdx.x;
  if (tid >= ETOT * HH) return;
  int e = tid >> 2, h = tid & 3;
  int dst = (e < EE) ? ei[EE + e] : (e - EE);
  float mx = ord2f(smaxI[dst * HH + h]);
  float ev = __expf(s[tid] - mx);
  s[tid] = ev;
  atomicAdd(denom + dst * HH + h, ev);
}

__global__ void edge_aggregate_kernel(const float* __restrict__ xl, const float* __restrict__ s,
                                      const float* __restrict__ denom, const int* __restrict__ ei,
                                      float* __restrict__ O, int D, int HD) {
  int tid = blockIdx.x * blockDim.x + threadIdx.x;
  if (tid >= ETOT * HH) return;
  int e = tid >> 2, h = tid & 3;
  int src, dst;
  if (e < EE) { src = ei[e]; dst = ei[EE + e]; } else { src = dst = e - EE; }
  float alpha = s[tid] / (denom[dst * HH + h] + 1e-16f);
  const float* pl = xl + (size_t)src * HD + h * D;
  float* po = O + (size_t)dst * HD + h * D;
  for (int d = 0; d < D; ++d)
    atomicAdd(po + d, alpha * pl[d]);
}

// ---------------- GraphNorm (3 passes, in place) ----------------
__global__ void gn_sum_kernel(const float* __restrict__ X, const int* __restrict__ batch,
                              float* __restrict__ gsum, int shift, int n) {
  int t = blockIdx.x * blockDim.x + threadIdx.x;
  if (t >= n) return;
  int dim = 1 << shift;
  int i = t >> shift, f = t & (dim - 1);
  atomicAdd(&gsum[batch[i] * dim + f], X[t]);
}
__global__ void gn_center_kernel(float* __restrict__ X, const int* __restrict__ batch,
                                 const float* __restrict__ gsum, const float* __restrict__ cnt,
                                 const float* __restrict__ a, float* __restrict__ gvar,
                                 int shift, int n) {
  int t = blockIdx.x * blockDim.x + threadIdx.x;
  if (t >= n) return;
  int dim = 1 << shift;
  int i = t >> shift, f = t & (dim - 1);
  int g = batch[i];
  float mean = gsum[g * dim + f] / cnt[g];
  float xc = X[t] - a[f] * mean;
  X[t] = xc;
  atomicAdd(&gvar[g * dim + f], xc * xc);
}
__global__ void gn_final_relu_kernel(float* __restrict__ X, const int* __restrict__ batch,
                                     const float* __restrict__ gvar, const float* __restrict__ cnt,
                                     const float* __restrict__ w, const float* __restrict__ b,
                                     int shift, int n) {
  int t = blockIdx.x * blockDim.x + threadIdx.x;
  if (t >= n) return;
  int dim = 1 << shift;
  int i = t >> shift, f = t & (dim - 1);
  int g = batch[i];
  float var = gvar[g * dim + f] / cnt[g];
  float y = w[f] * X[t] * rsqrtf(var + EPS_GN) + b[f];
  X[t] = fmaxf(y, 0.f);
}

// ---------------- layer-3 head mean + final pooling / linear ----------------
__global__ void head_mean_kernel(const float* __restrict__ O, const float* __restrict__ bias3,
                                 float* __restrict__ h3) {
  int t = blockIdx.x * blockDim.x + threadIdx.x;
  if (t >= NN * 16) return;
  int i = t >> 4, f = t & 15;
  const float* p = O + (size_t)i * 64;
  float sum = p[f] + p[16 + f] + p[32 + f] + p[48 + f];
  h3[t] = 0.25f * sum + bias3[f];
}
__global__ void pool_sum_kernel(const float* __restrict__ h3, const int* __restrict__ batch,
                                float* __restrict__ pooled) {
  int t = blockIdx.x * blockDim.x + threadIdx.x;
  if (t >= NN * 16) return;
  int i = t >> 4, f = t & 15;
  atomicAdd(&pooled[batch[i] * 16 + f], h3[t]);
}
__global__ void pool_final_kernel(const float* __restrict__ pooled, const float* __restrict__ cnt,
                                  float* __restrict__ out_features) {
  int t = blockIdx.x * blockDim.x + threadIdx.x;
  if (t >= GG * 16) return;
  int g = t >> 4;
  out_features[t] = pooled[t] / cnt[g];
}
__global__ void logits_kernel(const float* __restrict__ features, const float* __restrict__ linW,
                              const float* __restrict__ linB, float* __restrict__ out_logits) {
  int t = blockIdx.x * blockDim.x + threadIdx.x;
  if (t >= GG * CC) return;
  int g = t / CC, c = t - g * CC;
  float acc = linB[c];
  for (int f = 0; f < 16; ++f)
    acc += features[g * 16 + f] * linW[f * CC + c];
  out_logits[t] = acc;
}

// ---------------- host-side helpers ----------------
static inline int cdiv(int a, int b) { return (a + b - 1) / b; }

static void launch_gemm(const unsigned short* Xb, const unsigned short* Wp, const float* bias,
                        float* OUT, int K, int Nout, hipStream_t s) {
  dim3 grid(NN / 16, Nout / (16 * WPB)), blk(32 * WPB);
  if (K == 128)
    hipLaunchKernelGGL((gat_gemm_bf16_wmma<128>), grid, blk, 0, s, Xb, Wp, bias, OUT, Nout);
  else
    hipLaunchKernelGGL((gat_gemm_bf16_wmma<256>), grid, blk, 0, s, Xb, Wp, bias, OUT, Nout);
}
static void launch_cvt(const float* X, unsigned short* Xb, int nElem, hipStream_t s) {
  int n8 = nElem / 8;
  hipLaunchKernelGGL(cvt_bf16_kernel, dim3(cdiv(n8, 256)), dim3(256), 0, s, X, Xb, n8);
}
static void launch_pack(const float* W, unsigned short* Wp, int K, int Nout, hipStream_t s) {
  int total = (Nout / 16) * (K / 32) * 32;
  hipLaunchKernelGGL(pack_w_kernel, dim3(cdiv(total, 256)), dim3(256), 0, s, W, Wp, K, Nout);
}
static void run_graphnorm_relu(float* X, int shift, const int* batch, const float* cnt,
                               const float* w, const float* b, const float* a,
                               float* gsum, float* gvar, hipStream_t s) {
  int dim = 1 << shift;
  int n = NN * dim;
  hipLaunchKernelGGL(fill_f32_kernel, dim3(cdiv(GG * dim, 256)), dim3(256), 0, s, gsum, 0.f, GG * dim);
  hipLaunchKernelGGL(fill_f32_kernel, dim3(cdiv(GG * dim, 256)), dim3(256), 0, s, gvar, 0.f, GG * dim);
  hipLaunchKernelGGL(gn_sum_kernel, dim3(cdiv(n, 256)), dim3(256), 0, s, X, batch, gsum, shift, n);
  hipLaunchKernelGGL(gn_center_kernel, dim3(cdiv(n, 256)), dim3(256), 0, s, X, batch, gsum, cnt, a, gvar, shift, n);
  hipLaunchKernelGGL(gn_final_relu_kernel, dim3(cdiv(n, 256)), dim3(256), 0, s, X, batch, gvar, cnt, w, b, shift, n);
}
static void run_edge_phase(const float* xl, const float* xr, const float* att, const int* ei,
                           float* sE, int* smaxI, float* denom, float* O, int D, int HD,
                           hipStream_t s) {
  int tot = ETOT * HH;
  hipLaunchKernelGGL(fill_i32_kernel, dim3(cdiv(NN * HH, 256)), dim3(256), 0, s, smaxI, (int)0x80000000, NN * HH);
  hipLaunchKernelGGL(fill_f32_kernel, dim3(cdiv(NN * HH, 256)), dim3(256), 0, s, denom, 0.f, NN * HH);
  hipLaunchKernelGGL(edge_score_kernel, dim3(cdiv(tot, 256)), dim3(256), 0, s, xl, xr, att, ei, sE, smaxI, D, HD);
  hipLaunchKernelGGL(edge_softmax_kernel, dim3(cdiv(tot, 256)), dim3(256), 0, s, ei, smaxI, sE, denom);
  hipLaunchKernelGGL(edge_aggregate_kernel, dim3(cdiv(tot, 256)), dim3(256), 0, s, xl, sE, denom, ei, O, D, HD);
}

// ---------------- entry point ----------------
extern "C" void kernel_launch(void* const* d_in, const int* in_sizes, int n_in,
                              void* d_out, int out_size, void* d_ws, size_t ws_size,
                              hipStream_t stream) {
  (void)in_sizes; (void)n_in; (void)out_size; (void)ws_size;
  // inputs in setup_inputs() order
  const float* x     = (const float*)d_in[0];
  const int*   ei    = (const int*)d_in[1];   // [2,E]
  const int*   batch = (const int*)d_in[2];
  const float* Wl1 = (const float*)d_in[3];  const float* bl1 = (const float*)d_in[4];
  const float* Wr1 = (const float*)d_in[5];  const float* br1 = (const float*)d_in[6];
  const float* att1 = (const float*)d_in[7]; const float* bias1 = (const float*)d_in[8];
  const float* gw1 = (const float*)d_in[9];  const float* gb1 = (const float*)d_in[10];
  const float* ga1 = (const float*)d_in[11];
  const float* Wl2 = (const float*)d_in[12]; const float* bl2 = (const float*)d_in[13];
  const float* Wr2 = (const float*)d_in[14]; const float* br2 = (const float*)d_in[15];
  const float* att2 = (const float*)d_in[16]; const float* bias2 = (const float*)d_in[17];
  const float* gw2 = (const float*)d_in[18]; const float* gb2 = (const float*)d_in[19];
  const float* ga2 = (const float*)d_in[20];
  const float* Wl3 = (const float*)d_in[21]; const float* bl3 = (const float*)d_in[22];
  const float* Wr3 = (const float*)d_in[23]; const float* br3 = (const float*)d_in[24];
  const float* att3 = (const float*)d_in[25]; const float* bias3 = (const float*)d_in[26];
  const float* gw3 = (const float*)d_in[27]; const float* gb3 = (const float*)d_in[28];
  const float* ga3 = (const float*)d_in[29];
  const float* linW = (const float*)d_in[30]; const float* linB = (const float*)d_in[31];
  float* out = (float*)d_out;                 // [logits G*C | features G*16]

  // ---- workspace carve (floats) ----
  float* ws = (float*)d_ws;
  size_t o = 0;
  float* xl    = ws + o; o += (size_t)NN * 256;
  float* xr    = ws + o; o += (size_t)NN * 256;
  float* O     = ws + o; o += (size_t)NN * 256;   // reused as h1 / h2 / layer3 pre-mean
  float* sE    = ws + o; o += (size_t)ETOT * HH;
  int*   smaxI = (int*)(ws + o); o += (size_t)NN * HH;
  float* denom = ws + o; o += (size_t)NN * HH;
  float* h3    = ws + o; o += (size_t)NN * 16;
  float* gsum  = ws + o; o += (size_t)GG * 256;
  float* gvar  = ws + o; o += (size_t)GG * 256;
  float* cnt   = ws + o; o += GG;
  float* pooled= ws + o; o += (size_t)GG * 16;
  unsigned short* Xb = (unsigned short*)(ws + o); o += (size_t)NN * 128;  // bf16 activations, N*256
  unsigned short* Wl1p = (unsigned short*)(ws + o);
  unsigned short* Wr1p = Wl1p + 128 * 256;
  unsigned short* Wl2p = Wr1p + 128 * 256;
  unsigned short* Wr2p = Wl2p + 256 * 128;
  unsigned short* Wl3p = Wr2p + 256 * 128;
  unsigned short* Wr3p = Wl3p + 128 * 64;

  // ---- pack weights to bf16 fragment order ----
  launch_pack(Wl1, Wl1p, 128, 256, stream);
  launch_pack(Wr1, Wr1p, 128, 256, stream);
  launch_pack(Wl2, Wl2p, 256, 128, stream);
  launch_pack(Wr2, Wr2p, 256, 128, stream);
  launch_pack(Wl3, Wl3p, 128, 64, stream);
  launch_pack(Wr3, Wr3p, 128, 64, stream);

  // ---- per-graph node counts (clamped at 1) ----
  hipLaunchKernelGGL(fill_f32_kernel, dim3(1), dim3(256), 0, stream, cnt, 0.f, GG);
  hipLaunchKernelGGL(count_nodes_kernel, dim3(cdiv(NN, 256)), dim3(256), 0, stream, batch, cnt, NN);
  hipLaunchKernelGGL(clamp1_kernel, dim3(1), dim3(256), 0, stream, cnt, GG);

  // ---- layer 1: 128 -> 4x64 concat ----
  launch_cvt(x, Xb, NN * 128, stream);
  launch_gemm(Xb, Wl1p, bl1, xl, 128, 256, stream);
  launch_gemm(Xb, Wr1p, br1, xr, 128, 256, stream);
  hipLaunchKernelGGL(bias_bcast_kernel, dim3(cdiv(NN * 256, 256)), dim3(256), 0, stream, O, bias1, 255, NN * 256);
  run_edge_phase(xl, xr, att1, ei, sE, smaxI, denom, O, 64, 256, stream);
  run_graphnorm_relu(O, 8, batch, cnt, gw1, gb1, ga1, gsum, gvar, stream);

  // ---- layer 2: 256 -> 4x32 concat (reads O, then reuses O as output) ----
  launch_cvt(O, Xb, NN * 256, stream);
  launch_gemm(Xb, Wl2p, bl2, xl, 256, 128, stream);
  launch_gemm(Xb, Wr2p, br2, xr, 256, 128, stream);
  hipLaunchKernelGGL(bias_bcast_kernel, dim3(cdiv(NN * 128, 256)), dim3(256), 0, stream, O, bias2, 127, NN * 128);
  run_edge_phase(xl, xr, att2, ei, sE, smaxI, denom, O, 32, 128, stream);
  run_graphnorm_relu(O, 7, batch, cnt, gw2, gb2, ga2, gsum, gvar, stream);

  // ---- layer 3: 128 -> 4x16, heads averaged ----
  launch_cvt(O, Xb, NN * 128, stream);
  launch_gemm(Xb, Wl3p, bl3, xl, 128, 64, stream);
  launch_gemm(Xb, Wr3p, br3, xr, 128, 64, stream);
  hipLaunchKernelGGL(fill_f32_kernel, dim3(cdiv(NN * 64, 256)), dim3(256), 0, stream, O, 0.f, NN * 64);
  run_edge_phase(xl, xr, att3, ei, sE, smaxI, denom, O, 16, 64, stream);
  hipLaunchKernelGGL(head_mean_kernel, dim3(cdiv(NN * 16, 256)), dim3(256), 0, stream, O, bias3, h3);
  run_graphnorm_relu(h3, 4, batch, cnt, gw3, gb3, ga3, gsum, gvar, stream);

  // ---- global mean pool + final linear ----
  hipLaunchKernelGGL(fill_f32_kernel, dim3(cdiv(GG * 16, 256)), dim3(256), 0, stream, pooled, 0.f, GG * 16);
  hipLaunchKernelGGL(pool_sum_kernel, dim3(cdiv(NN * 16, 256)), dim3(256), 0, stream, h3, batch, pooled);
  hipLaunchKernelGGL(pool_final_kernel, dim3(1), dim3(512), 0, stream, pooled, cnt, out + GG * CC);
  hipLaunchKernelGGL(logits_kernel, dim3(1), dim3(128), 0, stream, out + GG * CC, linW, linB, out);
}
